// PATHQModel_45938970198549
// MI455X (gfx1250) — compile-verified, matrix operand order checked by reading, not run on hardware
//
#include <hip/hip_runtime.h>
#include <hip/hip_bf16.h>
#include <math.h>

typedef __attribute__((ext_vector_type(16))) _Float16 v16h;
typedef __attribute__((ext_vector_type(8)))  _Float16 v8h;
typedef __attribute__((ext_vector_type(8)))  float    v8f;

#define N_NODES 65536
#define N_EDGES 524288
#define N_BAGS  8
#define D_IN    512
#define H_DIM   256
#define HA_DIM  128
#define C_DIM   2
#define BN_EPS  1e-5f
#define MREP    4   // 4 m-tiles (64 rows) per wave, reuse B fragment

// ---------------------------------------------------------------- utilities
__global__ void fill_f32(float* p, float v, int n) {
    int t = blockIdx.x * blockDim.x + threadIdx.x;
    if (t < n) p[t] = v;
}

__global__ void deg_count(const int* __restrict__ dst, float* deg, int E) {
    int t = blockIdx.x * blockDim.x + threadIdx.x;
    if (t < E) atomicAdd(&deg[dst[t]], 1.0f);
}

__global__ void deg_to_dinv(float* deg, int N) {
    int t = blockIdx.x * blockDim.x + threadIdx.x;
    if (t < N) deg[t] = rsqrtf(deg[t]);   // deg was initialized to 1 (self loop)
}

// -------------------------------------------------- W pre-pack (B fragments)
// Pack W[K x Nout] (f32, row-major) into per-wave WMMA B fragments (f16):
//   out[ ((ntile*kblks + kblk)*32 + lane)*16 + i ]
// lane<16: col=ntile*16+lane, K = kblk*32 + {0..7, 16..23}
// lane>=16: same col group, K = kblk*32 + {8..15, 24..31}
__global__ void pack_w_f16(const float* __restrict__ W, _Float16* __restrict__ out,
                           int K, int Nout) {
    int t = blockIdx.x * blockDim.x + threadIdx.x;
    if (t >= K * Nout) return;
    int i    = t & 15;
    int lane = (t >> 4) & 31;
    int rest = t >> 9;
    int kblks = K >> 5;
    int ntile = rest / kblks;
    int kblk  = rest % kblks;
    int half = lane >> 4, l15 = lane & 15;
    int col = ntile * 16 + l15;
    int k   = kblk * 32 + half * 8 + ((i < 8) ? i : (16 + i - 8));
    out[t] = (_Float16)W[(size_t)k * Nout + col];
}

// ------------------------------------------------- WMMA GEMM, A = f32 global
// C[M x Nout] = A[M x K] @ W (pre-packed f16 B fragments). f32 accumulate.
__global__ void wmma_gemm_af32(const float* __restrict__ A,
                               const _Float16* __restrict__ Wpk,
                               float* __restrict__ C,
                               int M, int K, int Nout) {
    const int lane   = threadIdx.x & 31;
    const int ntiles = Nout >> 4;
    const int waveId = (blockIdx.x * blockDim.x + threadIdx.x) >> 5;
    const int mblk   = waveId / ntiles;
    const int ntile  = waveId % ntiles;
    const int m0     = mblk * (16 * MREP);
    if (m0 >= M) return;                         // wave-uniform exit

    const int half = lane >> 4;
    const int l15  = lane & 15;
    const int kblks = K >> 5;
    const _Float16* wbase = Wpk + ((size_t)ntile * kblks * 512) + lane * 16;

    v8f zero = {};
    v8f acc[MREP];
#pragma unroll
    for (int r = 0; r < MREP; ++r) acc[r] = zero;

    for (int kblk = 0; kblk < kblks; ++kblk) {
        v16h b = *(const v16h*)(wbase + (size_t)kblk * 512);
#pragma unroll
        for (int r = 0; r < MREP; ++r) {
            const float* ap = A + (size_t)(m0 + r * 16 + l15) * K + kblk * 32 + half * 8;
            v16h a;
#pragma unroll
            for (int i = 0; i < 8; ++i) a[i]     = (_Float16)ap[i];
#pragma unroll
            for (int i = 0; i < 8; ++i) a[8 + i] = (_Float16)ap[16 + i];
            acc[r] = __builtin_amdgcn_wmma_f32_16x16x32_f16(
                false, a, false, b, (short)0, acc[r], false, false);
        }
    }
#pragma unroll
    for (int r = 0; r < MREP; ++r) {
#pragma unroll
        for (int q = 0; q < 8; ++q) {
            const int row = m0 + r * 16 + q + half * 8;
            C[(size_t)row * Nout + (ntile * 16 + l15)] = acc[r][q];
        }
    }
}

// ------------------------------------------------- WMMA GEMM, A = f16 global
__global__ void wmma_gemm_af16(const _Float16* __restrict__ A,
                               const _Float16* __restrict__ Wpk,
                               const float* __restrict__ bias,
                               float* __restrict__ C,
                               int M, int K, int Nout, int addBias) {
    const int lane   = threadIdx.x & 31;
    const int ntiles = Nout >> 4;
    const int waveId = (blockIdx.x * blockDim.x + threadIdx.x) >> 5;
    const int mblk   = waveId / ntiles;
    const int ntile  = waveId % ntiles;
    const int m0     = mblk * (16 * MREP);
    if (m0 >= M) return;

    const int half = lane >> 4;
    const int l15  = lane & 15;
    const int colB = ntile * 16 + l15;
    const int kblks = K >> 5;
    const _Float16* wbase = Wpk + ((size_t)ntile * kblks * 512) + lane * 16;

    v8f zero = {};
    v8f acc[MREP];
#pragma unroll
    for (int r = 0; r < MREP; ++r) acc[r] = zero;

    for (int kblk = 0; kblk < kblks; ++kblk) {
        v16h b = *(const v16h*)(wbase + (size_t)kblk * 512);
#pragma unroll
        for (int r = 0; r < MREP; ++r) {
            const _Float16* ap = A + (size_t)(m0 + r * 16 + l15) * K + kblk * 32 + half * 8;
            v8h lo = *(const v8h*)(ap);
            v8h hi = *(const v8h*)(ap + 16);
            v16h a = __builtin_shufflevector(lo, hi, 0, 1, 2, 3, 4, 5, 6, 7,
                                             8, 9, 10, 11, 12, 13, 14, 15);
            acc[r] = __builtin_amdgcn_wmma_f32_16x16x32_f16(
                false, a, false, b, (short)0, acc[r], false, false);
        }
    }

    const float bv = addBias ? bias[colB] : 0.0f;
#pragma unroll
    for (int r = 0; r < MREP; ++r) {
#pragma unroll
        for (int q = 0; q < 8; ++q) {
            const int row = m0 + r * 16 + q + half * 8;
            C[(size_t)row * Nout + colB] = acc[r][q] + bv;
        }
    }
}

// ---------------------------------------------------------- message passing
__global__ void scatter_edges(const int* __restrict__ src,
                              const int* __restrict__ dst,
                              const float* __restrict__ dinv,
                              const float* __restrict__ h,
                              float* __restrict__ agg, int E, int H) {
    const int chunks = H >> 2;
    int t = blockIdx.x * blockDim.x + threadIdx.x;
    int e = t / chunks;
    if (e >= E) return;
    int f4 = (t % chunks) << 2;
    int s = src[e], d = dst[e];
    float coef = dinv[s] * dinv[d];
    const float4 hv = *(const float4*)(h + (size_t)s * H + f4);
    float* ag = agg + (size_t)d * H + f4;
    atomicAdd(ag + 0, hv.x * coef);
    atomicAdd(ag + 1, hv.y * coef);
    atomicAdd(ag + 2, hv.z * coef);
    atomicAdd(ag + 3, hv.w * coef);
}

// self-loop + conv bias + BN + ReLU; emits f16 (GEMM A-side) and optional f32
__global__ void gcn_finalize(const float* __restrict__ hlin,
                             const float* __restrict__ agg,
                             const float* __restrict__ dinv,
                             const float* __restrict__ bconv,
                             const float* __restrict__ g,
                             const float* __restrict__ bb,
                             const float* __restrict__ m,
                             const float* __restrict__ v,
                             _Float16* __restrict__ out16,
                             float* __restrict__ out32,   // may be null
                             int N, int H) {
    int idx = blockIdx.x * blockDim.x + threadIdx.x;
    if (idx >= N * H) return;
    int n = idx / H, f = idx - n * H;
    float di = dinv[n];
    float val = agg[idx] + hlin[idx] * di * di + bconv[f];
    val = (val - m[f]) * rsqrtf(v[f] + BN_EPS) * g[f] + bb[f];
    val = fmaxf(val, 0.0f);
    out16[idx] = (_Float16)val;
    if (out32) out32[idx] = val;
}

// ------------------------------------------------------------- attention
__global__ void attn_score(const float* __restrict__ av,
                           const float* __restrict__ au,
                           const float* __restrict__ wA,
                           float* __restrict__ a, int N, int HA) {
    int wave = (blockIdx.x * blockDim.x + threadIdx.x) >> 5;
    int lane = threadIdx.x & 31;
    if (wave >= N) return;
    float sum = 0.0f;
    for (int j = lane; j < HA; j += 32) {
        float t = tanhf(av[(size_t)wave * HA + j]);
        float s = 1.0f / (1.0f + expf(-au[(size_t)wave * HA + j]));
        sum += t * s * wA[j];
    }
    for (int off = 16; off > 0; off >>= 1) sum += __shfl_down(sum, off, 32);
    if (lane == 0) a[wave] = sum;
}

__global__ void seg_init(int* seg, int cnt, int sentinel) {
    int t = blockIdx.x * blockDim.x + threadIdx.x;
    if (t < cnt) seg[t] = sentinel;
}
__global__ void seg_mark(const int* __restrict__ batch, int* seg, int N) {
    int t = blockIdx.x * blockDim.x + threadIdx.x;
    if (t < N) atomicMin(&seg[batch[t]], t);
}
__global__ void seg_fix(int* seg, int B, int Nval) {
    if (blockIdx.x == 0 && threadIdx.x == 0) {
        seg[B] = Nval;
        for (int g = B - 1; g >= 0; --g)
            if (seg[g] == Nval) seg[g] = seg[g + 1];
    }
}

__global__ void seg_softmax(const float* __restrict__ a,
                            const int* __restrict__ seg,
                            float* __restrict__ wout) {
    __shared__ float red[256];
    int g = blockIdx.x;
    int s = seg[g], e = seg[g + 1];
    float mx = -1e30f;
    for (int n = s + (int)threadIdx.x; n < e; n += blockDim.x)
        mx = fmaxf(mx, a[n]);
    red[threadIdx.x] = mx; __syncthreads();
    for (int off = 128; off > 0; off >>= 1) {
        if ((int)threadIdx.x < off)
            red[threadIdx.x] = fmaxf(red[threadIdx.x], red[threadIdx.x + off]);
        __syncthreads();
    }
    mx = red[0]; __syncthreads();
    float sum = 0.0f;
    for (int n = s + (int)threadIdx.x; n < e; n += blockDim.x)
        sum += expf(a[n] - mx);
    red[threadIdx.x] = sum; __syncthreads();
    for (int off = 128; off > 0; off >>= 1) {
        if ((int)threadIdx.x < off)
            red[threadIdx.x] += red[threadIdx.x + off];
        __syncthreads();
    }
    float inv = 1.0f / red[0];
    for (int n = s + (int)threadIdx.x; n < e; n += blockDim.x)
        wout[n] = expf(a[n] - mx) * inv;
}

__global__ void slide_acc(const float* __restrict__ w,
                          const float* __restrict__ h,
                          const int* __restrict__ seg,
                          float* __restrict__ slide, int H, int chunks) {
    int g = blockIdx.x / chunks;
    int c = blockIdx.x % chunks;
    int s = seg[g], e = seg[g + 1];
    int len = e - s;
    int per = (len + chunks - 1) / chunks;
    int n0 = s + c * per;
    int n1 = min(n0 + per, e);
    int f = threadIdx.x;
    float acc = 0.0f;
    for (int n = n0; n < n1; ++n) acc += w[n] * h[(size_t)n * H + f];
    if (n0 < n1) atomicAdd(&slide[g * H + f], acc);
}

__global__ void head_kernel(const float* __restrict__ slide,
                            const float* __restrict__ Wc1,
                            const float* __restrict__ bc1,
                            const float* __restrict__ Wc2,
                            const float* __restrict__ bc2,
                            float* __restrict__ logits) {
    __shared__ float hid[N_BAGS * HA_DIM];
    for (int item = threadIdx.x; item < N_BAGS * HA_DIM; item += blockDim.x) {
        int g = item / HA_DIM, t = item - g * HA_DIM;
        float s = bc1[t];
        for (int k = 0; k < H_DIM; ++k) s += slide[g * H_DIM + k] * Wc1[k * HA_DIM + t];
        hid[item] = fmaxf(s, 0.0f);
    }
    __syncthreads();
    for (int item = threadIdx.x; item < N_BAGS * C_DIM; item += blockDim.x) {
        int g = item / C_DIM, c = item - g * C_DIM;
        float s = bc2[c];
        for (int t = 0; t < HA_DIM; ++t) s += hid[g * HA_DIM + t] * Wc2[t * C_DIM + c];
        logits[item] = s;
    }
}

// ---------------------------------------------------------------- launcher
static inline int cdiv(long long a, int b) { return (int)((a + b - 1) / b); }

extern "C" void kernel_launch(void* const* d_in, const int* in_sizes, int n_in,
                              void* d_out, int out_size, void* d_ws, size_t ws_size,
                              hipStream_t stream) {
    (void)in_sizes; (void)n_in; (void)out_size; (void)ws_size;
    const float* x   = (const float*)d_in[0];
    const int*  ei   = (const int*)d_in[1];
    const int*  batch= (const int*)d_in[2];
    const float* W1  = (const float*)d_in[3];  const float* b1   = (const float*)d_in[4];
    const float* g1  = (const float*)d_in[5];  const float* bb1  = (const float*)d_in[6];
    const float* m1  = (const float*)d_in[7];  const float* v1   = (const float*)d_in[8];
    const float* W2  = (const float*)d_in[9];  const float* b2   = (const float*)d_in[10];
    const float* g2  = (const float*)d_in[11]; const float* bb2  = (const float*)d_in[12];
    const float* m2  = (const float*)d_in[13]; const float* v2   = (const float*)d_in[14];
    const float* Wv  = (const float*)d_in[15]; const float* bv   = (const float*)d_in[16];
    const float* Wu  = (const float*)d_in[17]; const float* bu   = (const float*)d_in[18];
    const float* wA  = (const float*)d_in[19];
    const float* Wc1 = (const float*)d_in[20]; const float* bc1  = (const float*)d_in[21];
    const float* Wc2 = (const float*)d_in[22]; const float* bc2  = (const float*)d_in[23];

    const int* src = ei;
    const int* dst = ei + N_EDGES;

    float* out  = (float*)d_out;             // [0..15] logits, [16..] w
    float* wout = out + N_BAGS * C_DIM;

    const size_t NH = (size_t)N_NODES * H_DIM;
    float*    bufA = (float*)d_ws;                         // N x H f32
    float*    bufB = bufA + NH;                            // N x H f32
    _Float16* h16  = (_Float16*)(bufB + NH);               // N x H f16
    _Float16* w1pk = h16 + NH;                             // 131072 f16
    _Float16* w2pk = w1pk + (size_t)D_IN * H_DIM;          // 65536 f16
    _Float16* wvpk = w2pk + (size_t)H_DIM * H_DIM;         // 32768 f16
    _Float16* wupk = wvpk + (size_t)H_DIM * HA_DIM;        // 32768 f16
    float*    dinv = (float*)(wupk + (size_t)H_DIM * HA_DIM);
    float*    abuf = dinv + N_NODES;
    float*    slide= abuf + N_NODES;
    int*      seg  = (int*)(slide + N_BAGS * H_DIM);

    const int TB = 256;
    const int NHb = cdiv((long long)NH, TB);

    // ---- degree / dinv
    fill_f32<<<cdiv(N_NODES, TB), TB, 0, stream>>>(dinv, 1.0f, N_NODES);
    deg_count<<<cdiv(N_EDGES, TB), TB, 0, stream>>>(dst, dinv, N_EDGES);
    deg_to_dinv<<<cdiv(N_NODES, TB), TB, 0, stream>>>(dinv, N_NODES);

    // ---- pre-pack all weight matrices into WMMA B-fragment f16 layout
    pack_w_f16<<<cdiv((long long)D_IN * H_DIM, TB), TB, 0, stream>>>(W1, w1pk, D_IN, H_DIM);
    pack_w_f16<<<cdiv((long long)H_DIM * H_DIM, TB), TB, 0, stream>>>(W2, w2pk, H_DIM, H_DIM);
    pack_w_f16<<<cdiv((long long)H_DIM * HA_DIM, TB), TB, 0, stream>>>(Wv, wvpk, H_DIM, HA_DIM);
    pack_w_f16<<<cdiv((long long)H_DIM * HA_DIM, TB), TB, 0, stream>>>(Wu, wupk, H_DIM, HA_DIM);

    // ---- GCN layer 1: hlin1 = x @ W1 -> bufA
    {
        long long waves = ((long long)N_NODES / 64) * (H_DIM / 16);
        wmma_gemm_af32<<<cdiv(waves * 32, TB), TB, 0, stream>>>(
            x, w1pk, bufA, N_NODES, D_IN, H_DIM);
    }
    fill_f32<<<NHb, TB, 0, stream>>>(bufB, 0.0f, (int)NH);
    scatter_edges<<<cdiv((long long)N_EDGES * (H_DIM / 4), TB), TB, 0, stream>>>(
        src, dst, dinv, bufA, bufB, N_EDGES, H_DIM);
    gcn_finalize<<<NHb, TB, 0, stream>>>(bufA, bufB, dinv, b1, g1, bb1, m1, v1,
                                         h16, nullptr, N_NODES, H_DIM);   // h1 -> h16

    // ---- GCN layer 2: hlin2 = h1 @ W2 -> bufA
    {
        long long waves = ((long long)N_NODES / 64) * (H_DIM / 16);
        wmma_gemm_af16<<<cdiv(waves * 32, TB), TB, 0, stream>>>(
            h16, w2pk, nullptr, bufA, N_NODES, H_DIM, H_DIM, 0);
    }
    fill_f32<<<NHb, TB, 0, stream>>>(bufB, 0.0f, (int)NH);
    scatter_edges<<<cdiv((long long)N_EDGES * (H_DIM / 4), TB), TB, 0, stream>>>(
        src, dst, dinv, bufA, bufB, N_EDGES, H_DIM);
    gcn_finalize<<<NHb, TB, 0, stream>>>(bufA, bufB, dinv, b2, g2, bb2, m2, v2,
                                         h16, bufA, N_NODES, H_DIM);      // h2 -> h16 + bufA

    // ---- gated attention projections: av|au -> bufB
    float* av = bufB;
    float* au = bufB + (size_t)N_NODES * HA_DIM;
    {
        long long waves = ((long long)N_NODES / 64) * (HA_DIM / 16);
        wmma_gemm_af16<<<cdiv(waves * 32, TB), TB, 0, stream>>>(
            h16, wvpk, bv, av, N_NODES, H_DIM, HA_DIM, 1);
        wmma_gemm_af16<<<cdiv(waves * 32, TB), TB, 0, stream>>>(
            h16, wupk, bu, au, N_NODES, H_DIM, HA_DIM, 1);
    }
    attn_score<<<cdiv((long long)N_NODES * 32, TB), TB, 0, stream>>>(
        av, au, wA, abuf, N_NODES, HA_DIM);

    // ---- segment boundaries (batch sorted)
    seg_init<<<1, 32, 0, stream>>>(seg, N_BAGS + 1, N_NODES);
    seg_mark<<<cdiv(N_NODES, TB), TB, 0, stream>>>(batch, seg, N_NODES);
    seg_fix<<<1, 32, 0, stream>>>(seg, N_BAGS, N_NODES);

    // ---- per-bag softmax -> w written to d_out+16
    seg_softmax<<<N_BAGS, 256, 0, stream>>>(abuf, seg, wout);

    // ---- weighted pooling (h2 f32 in bufA) -> slide
    fill_f32<<<cdiv(N_BAGS * H_DIM, TB), TB, 0, stream>>>(slide, 0.0f, N_BAGS * H_DIM);
    const int CHUNKS = 16;
    slide_acc<<<N_BAGS * CHUNKS, H_DIM, 0, stream>>>(wout, bufA, seg, slide, H_DIM, CHUNKS);

    // ---- classifier head -> logits
    head_kernel<<<1, 256, 0, stream>>>(slide, Wc1, bc1, Wc2, bc2, out);
}
